// AttModel_self_syb_40235253629208
// MI455X (gfx1250) — matrix operand order repaired and checked by reading, not compiled, not run on hardware
//
#include <hip/hip_runtime.h>
#include <hip/hip_bf16.h>

// ---------------------------------------------------------------------------
// CDNA5 (gfx1250) wave32 WMMA implementation of the graph-attention model.
// All GEMMs use v_wmma_f32_16x16x32_bf16 with f32 accumulation.
// B-tiles are staged to LDS with GLOBAL_LOAD_ASYNC_TO_LDS_B128 (ASYNCcnt),
// double-buffered so the next tile's fetch overlaps the current tile's WMMAs.
// ---------------------------------------------------------------------------

typedef __bf16 bf16_t;
typedef __attribute__((ext_vector_type(16))) __bf16 v16bf;
typedef __attribute__((ext_vector_type(8)))  __bf16 v8bf;
typedef __attribute__((ext_vector_type(8)))  float  v8f;

#define B_    16
#define T_    256
#define D_    1024
#define H_    16
#define DH_   64
#define NB_   6
#define GD_   300
#define GDP_  320          // GD padded to multiple of 32 for WMMA K-dim
#define MLPH_ 2048
#define FFH_  4096
#define M_    (B_ * T_)    // 4096 rows
#define NEGV  (-4294967295.0f)

// Async global->LDS staging path (gfx1250 GLOBAL_LOAD_ASYNC_TO_LDS + ASYNCcnt)
#if __has_builtin(__builtin_amdgcn_global_load_async_to_lds_b128) && \
    __has_builtin(__builtin_amdgcn_s_wait_asynccnt)
#define HAVE_ASYNC_LDS 1
#else
#define HAVE_ASYNC_LDS 0
#endif

#if HAVE_ASYNC_LDS
// Exact param type per hipcc diagnostic: 'int __vector(4) __device__ *'
typedef int v4i_g __attribute__((vector_size(16)));
typedef __attribute__((address_space(1))) v4i_g* as1_v4i;
typedef __attribute__((address_space(3))) v4i_g* as3_v4i;

__device__ __forceinline__ void async_copy16(const bf16_t* gsrc, bf16_t* ldst) {
  __builtin_amdgcn_global_load_async_to_lds_b128(
      (as1_v4i)(void*)const_cast<bf16_t*>(gsrc), (as3_v4i)(void*)ldst, 0, 0);
}
#endif

// ---------------------------------------------------------------------------
// WMMA helpers (wave32).  Layouts per CDNA5 ISA 7.12.2:
//   A (16x32 bf16): lane l -> row M = l&15; element e -> K in two contiguous
//     8-element groups: K = 16*(e>>3) + 8*(l>>4) + (e&7)
//   B (32x16 bf16): lane l -> col N = l&15; element e -> K = 16*(e>>3) + 8*(l>>4) + (e&7)
//   C/D (16x16 f32): VGPR r, lane l -> row M = r + 8*(l>>4), col N = l&15
// ---------------------------------------------------------------------------

__device__ __forceinline__ v8f wmma_bf16(v16bf a, v16bf b, v8f c) {
  return __builtin_amdgcn_wmma_f32_16x16x32_bf16(
      /*neg_a=*/false, a, /*neg_b=*/false, b,
      /*c_mod=*/(short)0, c, /*reuse_a=*/false, /*reuse_b=*/false);
}

// A-fragment from row-major [M x ld] bf16 (ld multiple of 8; 16B-aligned rows)
__device__ __forceinline__ v16bf load_a_frag(const bf16_t* A, int ld, int m0, int k0) {
  int lane = threadIdx.x & 31;
  const bf16_t* p = A + (size_t)(m0 + (lane & 15)) * (size_t)ld + k0 + ((lane >> 4) << 3);
  v8bf lo = *(const v8bf*)(p);        // K = k0 + 8*half + [0..7]
  v8bf hi = *(const v8bf*)(p + 16);   // K = k0 + 16 + 8*half + [0..7]
  v16bf a;
#pragma unroll
  for (int e = 0; e < 8; ++e) { a[e] = lo[e]; a[8 + e] = hi[e]; }
  return a;
}

// B-fragment from K-major matrix BT [N x ld] bf16 (row n holds the K-dim)
__device__ __forceinline__ v16bf load_b_fragT(const bf16_t* BT, int ld, int n0, int k0) {
  int lane = threadIdx.x & 31;
  const bf16_t* p = BT + (size_t)(n0 + (lane & 15)) * (size_t)ld + k0 + ((lane >> 4) << 3);
  v8bf lo = *(const v8bf*)(p);        // K = k0 + 8*half + [0..7]
  v8bf hi = *(const v8bf*)(p + 16);   // K = k0 + 16 + 8*half + [0..7]
  v16bf b;
#pragma unroll
  for (int e = 0; e < 8; ++e) { b[e] = lo[e]; b[8 + e] = hi[e]; }
  return b;
}

// ---------------------------------------------------------------------------
// Kernel: gather embedding rows -> bf16, zero-pad 300 -> 320
// ---------------------------------------------------------------------------
__global__ void __launch_bounds__(256)
gather_embed_kernel(const int* __restrict__ ids, const float* __restrict__ emb,
                    bf16_t* __restrict__ X0) {
  size_t idx = (size_t)blockIdx.x * 256 + threadIdx.x;
  if (idx >= (size_t)M_ * GDP_) return;
  int row = (int)(idx / GDP_);
  int c   = (int)(idx % GDP_);
  int tok = ids[row];
  X0[idx] = (c < GD_) ? (bf16_t)emb[(size_t)tok * GD_ + c] : (bf16_t)0.0f;
}

// ---------------------------------------------------------------------------
// Kernel: convert fp32 weight [K x N] -> bf16 transposed [N x Kp] (zero-pad K)
// ---------------------------------------------------------------------------
__global__ void __launch_bounds__(256)
convT_kernel(const float* __restrict__ W, bf16_t* __restrict__ WT,
             int K, int N, int Kp) {
  size_t idx = (size_t)blockIdx.x * 256 + threadIdx.x;
  if (idx >= (size_t)N * Kp) return;
  int n = (int)(idx / Kp);
  int k = (int)(idx % Kp);
  WT[idx] = (k < K) ? (bf16_t)W[(size_t)k * N + n] : (bf16_t)0.0f;
}

// ---------------------------------------------------------------------------
// Kernel: generic WMMA GEMM  C[M,N] = act(A[M,Kp]_bf16 @ WT[N,Kp]_bf16 + bias)
// Block = 256 threads = 8 waves; tile 256(M) x 64(N); wave w owns two
// 16-row stripes (m0, m0+128) so each LDS B-fragment feeds 2 WMMAs.
// cb_trans: write bf16 output transposed (Cb[n*M + m]) -- used for V.
// ---------------------------------------------------------------------------
__global__ void __launch_bounds__(256)
gemm_bf16_kernel(const bf16_t* __restrict__ A, const bf16_t* __restrict__ WT,
                 const float* __restrict__ bias, float* __restrict__ Cf,
                 bf16_t* __restrict__ Cb, int M, int N, int Kp, int relu,
                 int cb_trans) {
  int w    = threadIdx.x >> 5;
  int lane = threadIdx.x & 31;
  int m0   = blockIdx.y * 256 + w * 16;
  int n0   = blockIdx.x * 64;

  v8f acc[2][4] = {};

#if HAVE_ASYNC_LDS
  __shared__ __align__(16) bf16_t Btile[2][64 * 32];
  int br  = threadIdx.x >> 2;                 // 0..63 : local n-row of tile
  int bs  = (threadIdx.x & 3) << 3;           // 16B segment within 64B row
  const bf16_t* gsrc = WT + (size_t)(n0 + br) * Kp + bs;
  bf16_t* ldst[2] = { &Btile[0][br * 32 + bs], &Btile[1][br * 32 + bs] };

  async_copy16(gsrc, ldst[0]);                // prefetch tile 0
  int nsteps = Kp >> 5;
  for (int ki = 0; ki < nsteps; ++ki) {
    int k = ki << 5;
    __syncthreads();      // everyone done reading the buffer we overwrite next
    if (ki + 1 < nsteps) {
      async_copy16(gsrc + k + 32, ldst[(ki + 1) & 1]);
      __builtin_amdgcn_s_wait_asynccnt(1);    // only tile ki must be complete
    } else {
      __builtin_amdgcn_s_wait_asynccnt(0);
    }
    __syncthreads();      // whole tile ki visible to all waves
    const bf16_t* bt = &Btile[ki & 1][0];
    v16bf a0 = load_a_frag(A, Kp, m0, k);
    v16bf a1 = load_a_frag(A, Kp, m0 + 128, k);
    if (k + 64 < Kp)
      __builtin_prefetch(A + (size_t)(m0 + (lane & 15)) * Kp + k + 64, 0, 3);
    v16bf b0 = load_b_fragT(bt, 32, 0,  0);
    v16bf b1 = load_b_fragT(bt, 32, 16, 0);
    v16bf b2 = load_b_fragT(bt, 32, 32, 0);
    v16bf b3 = load_b_fragT(bt, 32, 48, 0);
    acc[0][0] = wmma_bf16(a0, b0, acc[0][0]);
    acc[1][0] = wmma_bf16(a1, b0, acc[1][0]);
    acc[0][1] = wmma_bf16(a0, b1, acc[0][1]);
    acc[1][1] = wmma_bf16(a1, b1, acc[1][1]);
    acc[0][2] = wmma_bf16(a0, b2, acc[0][2]);
    acc[1][2] = wmma_bf16(a1, b2, acc[1][2]);
    acc[0][3] = wmma_bf16(a0, b3, acc[0][3]);
    acc[1][3] = wmma_bf16(a1, b3, acc[1][3]);
  }
#else
  for (int k = 0; k < Kp; k += 32) {
    v16bf a0 = load_a_frag(A, Kp, m0, k);
    v16bf a1 = load_a_frag(A, Kp, m0 + 128, k);
    v16bf b0 = load_b_fragT(WT, Kp, n0 +  0, k);
    v16bf b1 = load_b_fragT(WT, Kp, n0 + 16, k);
    v16bf b2 = load_b_fragT(WT, Kp, n0 + 32, k);
    v16bf b3 = load_b_fragT(WT, Kp, n0 + 48, k);
    if (k + 64 < Kp)
      __builtin_prefetch(A + (size_t)(m0 + (lane & 15)) * Kp + k + 64, 0, 3);
    acc[0][0] = wmma_bf16(a0, b0, acc[0][0]);
    acc[1][0] = wmma_bf16(a1, b0, acc[1][0]);
    acc[0][1] = wmma_bf16(a0, b1, acc[0][1]);
    acc[1][1] = wmma_bf16(a1, b1, acc[1][1]);
    acc[0][2] = wmma_bf16(a0, b2, acc[0][2]);
    acc[1][2] = wmma_bf16(a1, b2, acc[1][2]);
    acc[0][3] = wmma_bf16(a0, b3, acc[0][3]);
    acc[1][3] = wmma_bf16(a1, b3, acc[1][3]);
  }
#endif

  int half = lane >> 4, nc = lane & 15;
#pragma unroll
  for (int s = 0; s < 2; ++s) {
#pragma unroll
    for (int t = 0; t < 4; ++t) {
      int n = n0 + t * 16 + nc;
      float bv = bias ? bias[n] : 0.0f;
#pragma unroll
      for (int r = 0; r < 8; ++r) {
        int m = m0 + s * 128 + r + 8 * half;
        float v = acc[s][t][r] + bv;
        if (relu) v = fmaxf(v, 0.0f);
        if (Cf) Cf[(size_t)m * N + n] = v;
        if (Cb) {
          if (cb_trans) Cb[(size_t)n * M + m] = (bf16_t)v;   // feature-major (Vt)
          else          Cb[(size_t)m * N + n] = (bf16_t)v;
        }
      }
    }
  }
}

// ---------------------------------------------------------------------------
// Kernel: x += pos_table[t];  also emit bf16 copy
// ---------------------------------------------------------------------------
__global__ void __launch_bounds__(256)
add_pos_kernel(float* __restrict__ x, const float* __restrict__ pos,
               bf16_t* __restrict__ xb) {
  int row = blockIdx.x;            // 0..4095
  int t   = row & (T_ - 1);
  size_t base = (size_t)row * D_;
#pragma unroll
  for (int i = 0; i < 4; ++i) {
    int c = threadIdx.x + i * 256;
    float v = x[base + c] + pos[(size_t)t * D_ + c];
    x[base + c]  = v;
    xb[base + c] = (bf16_t)v;
  }
}

// ---------------------------------------------------------------------------
// Kernel: fused masked attention.
// Grid (T/32, H, B); block = 64 threads = 2 waves, each owning a 16-row stripe.
// S stripe (32x256 f32) and P (bf16 probs) live in LDS.  V is consumed in
// transposed (feature-major) layout Vt[D][M] so B-fragments are contiguous.
// ---------------------------------------------------------------------------
__global__ void __launch_bounds__(64)
attn_kernel(const bf16_t* __restrict__ Q, const bf16_t* __restrict__ K,
            const bf16_t* __restrict__ Vt, const int* __restrict__ kmask,
            const int* __restrict__ gmask, float* __restrict__ O) {
  __shared__ float  S[32][T_];
  __shared__ __align__(16) bf16_t P[32][264];   // padded so rows stay 16B-aligned

  int b = blockIdx.z, h = blockIdx.y, qt = blockIdx.x;
  int w = threadIdx.x >> 5, lane = threadIdx.x & 31;
  int half = lane >> 4, nc = lane & 15;

  const bf16_t* Qb  = Q  + ((size_t)b * T_) * D_ + h * DH_;
  const bf16_t* Kb  = K  + ((size_t)b * T_) * D_ + h * DH_;
  const bf16_t* Vtb = Vt + (size_t)(h * DH_) * M_ + (size_t)b * T_;
  int mrow = qt * 32 + w * 16;    // first query row (t index) of this wave

  // ---- S = (Q . K^T) / sqrt(dh) ----
  v16bf a0 = load_a_frag(Qb, D_, mrow, 0);
  v16bf a1 = load_a_frag(Qb, D_, mrow, 32);
#pragma unroll
  for (int ct = 0; ct < 16; ++ct) {
    v16bf b0 = load_b_fragT(Kb, D_, ct * 16, 0);
    v16bf b1 = load_b_fragT(Kb, D_, ct * 16, 32);
    v8f acc = {};
    acc = wmma_bf16(a0, b0, acc);
    acc = wmma_bf16(a1, b1, acc);
#pragma unroll
    for (int r = 0; r < 8; ++r)
      S[w * 16 + r + 8 * half][ct * 16 + nc] = acc[r] * 0.125f;  // 1/sqrt(64)
  }
  __syncthreads();

  // ---- masked softmax (one thread per query row) ----
  if (threadIdx.x < 32) {
    int r  = threadIdx.x;
    int qi = qt * 32 + r;
    const int* km = kmask + (size_t)b * T_;
    const int* gm = gmask + ((size_t)b * T_ + qi) * T_;
    float mx = -3.0e38f;
    for (int j = 0; j < T_; ++j) {
      float s = ((km[j] > 0) && (gm[j] > 0)) ? S[r][j] : NEGV;
      S[r][j] = s;
      mx = fmaxf(mx, s);
    }
    float sum = 0.0f;
    for (int j = 0; j < T_; ++j) {
      float e = __expf(S[r][j] - mx);
      S[r][j] = e;
      sum += e;
    }
    float inv = (float)km[qi] / sum;   // qscale * 1/sum
    for (int j = 0; j < T_; ++j) P[r][j] = (bf16_t)(S[r][j] * inv);
  }
  __syncthreads();

  // ---- O = P @ V ----
  v8f oacc[4] = {};
  for (int kk = 0; kk < T_; kk += 32) {
    v16bf pa = load_a_frag(&P[0][0], 264, w * 16, kk);
    v16bf b0 = load_b_fragT(Vtb, M_,  0, kk);
    v16bf b1 = load_b_fragT(Vtb, M_, 16, kk);
    v16bf b2 = load_b_fragT(Vtb, M_, 32, kk);
    v16bf b3 = load_b_fragT(Vtb, M_, 48, kk);
    oacc[0] = wmma_bf16(pa, b0, oacc[0]);
    oacc[1] = wmma_bf16(pa, b1, oacc[1]);
    oacc[2] = wmma_bf16(pa, b2, oacc[2]);
    oacc[3] = wmma_bf16(pa, b3, oacc[3]);
  }
#pragma unroll
  for (int t = 0; t < 4; ++t) {
#pragma unroll
    for (int r = 0; r < 8; ++r) {
      int m = b * T_ + qt * 32 + w * 16 + r + 8 * half;
      int n = h * DH_ + t * 16 + nc;
      O[(size_t)m * D_ + n] = oacc[t][r];
    }
  }
}

// ---------------------------------------------------------------------------
// Kernel: x = LayerNorm(a + x) * g + b;  also emit bf16 copy. One block/row.
// ---------------------------------------------------------------------------
__global__ void __launch_bounds__(256)
add_ln_kernel(const float* __restrict__ a, const float* __restrict__ xin,
              const float* __restrict__ g, const float* __restrict__ bt,
              float* __restrict__ xout, bf16_t* __restrict__ xb) {
  __shared__ float red[256];
  int tid = threadIdx.x;
  size_t base = (size_t)blockIdx.x * D_;

  float v[4];
  float s = 0.0f;
#pragma unroll
  for (int i = 0; i < 4; ++i) {
    int c = tid + i * 256;
    v[i] = a[base + c] + xin[base + c];
    s += v[i];
  }
  red[tid] = s;
  __syncthreads();
  for (int st = 128; st > 0; st >>= 1) {
    if (tid < st) red[tid] += red[tid + st];
    __syncthreads();
  }
  float mean = red[0] * (1.0f / D_);
  __syncthreads();

  float s2 = 0.0f;
#pragma unroll
  for (int i = 0; i < 4; ++i) { float d = v[i] - mean; s2 += d * d; }
  red[tid] = s2;
  __syncthreads();
  for (int st = 128; st > 0; st >>= 1) {
    if (tid < st) red[tid] += red[tid + st];
    __syncthreads();
  }
  float rstd = rsqrtf(red[0] * (1.0f / D_) + 1e-8f);

#pragma unroll
  for (int i = 0; i < 4; ++i) {
    int c = tid + i * 256;
    float o = (v[i] - mean) * rstd * g[c] + bt[c];
    xout[base + c] = o;
    xb[base + c]   = (bf16_t)o;
  }
}

// ---------------------------------------------------------------------------
// Host orchestration
// ---------------------------------------------------------------------------
extern "C" void kernel_launch(void* const* d_in, const int* in_sizes, int n_in,
                              void* d_out, int out_size, void* d_ws, size_t ws_size,
                              hipStream_t stream) {
  (void)in_sizes; (void)n_in; (void)out_size; (void)ws_size;

  const int*   syb_ipt   = (const int*)d_in[0];
  const int*   syb_mask  = (const int*)d_in[1];
  const int*   syb_graph = (const int*)d_in[2];
  const float* emb_table = (const float*)d_in[3];
  const float* mlp_w1    = (const float*)d_in[4];
  const float* mlp_b1    = (const float*)d_in[5];
  const float* mlp_w2    = (const float*)d_in[6];
  const float* mlp_b2    = (const float*)d_in[7];
  const float* pos_table = (const float*)d_in[8];
  const float* wq        = (const float*)d_in[9];
  const float* bq        = (const float*)d_in[10];
  const float* wk        = (const float*)d_in[11];
  const float* bk        = (const float*)d_in[12];
  const float* wv        = (const float*)d_in[13];
  const float* bv        = (const float*)d_in[14];
  const float* ln1_g     = (const float*)d_in[15];
  const float* ln1_b     = (const float*)d_in[16];
  const float* ff_w1     = (const float*)d_in[17];
  const float* ff_b1     = (const float*)d_in[18];
  const float* ff_w2     = (const float*)d_in[19];
  const float* ff_b2     = (const float*)d_in[20];
  const float* ln2_g     = (const float*)d_in[21];
  const float* ln2_b     = (const float*)d_in[22];

  char* ws = (char*)d_ws;
  size_t off = 0;
  auto carve = [&](size_t bytes) {
    void* p = ws + off;
    off = (off + bytes + 255) & ~(size_t)255;
    return p;
  };
  bf16_t* X0   = (bf16_t*)carve((size_t)M_ * GDP_ * 2);   //  2.6 MB
  bf16_t* xb   = (bf16_t*)carve((size_t)M_ * D_ * 2);     //  8.4 MB
  float*  xbuf = (float*) carve((size_t)M_ * D_ * 4);     // 16.8 MB
  bf16_t* hbig = (bf16_t*)carve((size_t)M_ * FFH_ * 2);   // 33.6 MB
  bf16_t* Qb   = (bf16_t*)carve((size_t)M_ * D_ * 2);
  bf16_t* Kb   = (bf16_t*)carve((size_t)M_ * D_ * 2);
  bf16_t* Vt   = (bf16_t*)carve((size_t)D_ * M_ * 2);     // feature-major V
  float*  obuf = (float*) carve((size_t)M_ * D_ * 4);     // 16.8 MB
  bf16_t* wT   = (bf16_t*)carve((size_t)FFH_ * D_ * 2);   //  8.4 MB (max weight)

  auto convT = [&](const float* W, int K, int N, int Kp) {
    size_t n = (size_t)N * Kp;
    convT_kernel<<<dim3((unsigned)((n + 255) / 256)), dim3(256), 0, stream>>>(W, wT, K, N, Kp);
  };
  auto gemm = [&](const bf16_t* A, const float* bias, float* Cf, bf16_t* Cb,
                  int N, int Kp, int relu, int cb_trans) {
    gemm_bf16_kernel<<<dim3(N / 64, M_ / 256), dim3(256), 0, stream>>>(
        A, wT, bias, Cf, Cb, M_, N, Kp, relu, cb_trans);
  };

  // --- embedding + front-end MLP ---
  {
    size_t n = (size_t)M_ * GDP_;
    gather_embed_kernel<<<dim3((unsigned)((n + 255) / 256)), dim3(256), 0, stream>>>(
        syb_ipt, emb_table, X0);
  }
  convT(mlp_w1, GD_, MLPH_, GDP_);
  gemm(X0, mlp_b1, nullptr, hbig, MLPH_, GDP_, 1, 0);        // h0 = relu(X0 @ W1 + b1)
  convT(mlp_w2, MLPH_, D_, MLPH_);
  gemm(hbig, mlp_b2, xbuf, nullptr, D_, MLPH_, 0, 0);        // x = h0 @ W2 + b2
  add_pos_kernel<<<dim3(M_), dim3(256), 0, stream>>>(xbuf, pos_table, xb);

  // --- transformer blocks ---
  for (int i = 0; i < NB_; ++i) {
    convT(wq + (size_t)i * D_ * D_, D_, D_, D_);
    gemm(xb, bq + (size_t)i * D_, nullptr, Qb, D_, D_, 1, 0);
    convT(wk + (size_t)i * D_ * D_, D_, D_, D_);
    gemm(xb, bk + (size_t)i * D_, nullptr, Kb, D_, D_, 1, 0);
    convT(wv + (size_t)i * D_ * D_, D_, D_, D_);
    gemm(xb, bv + (size_t)i * D_, nullptr, Vt, D_, D_, 1, 1);  // transposed bf16 out

    attn_kernel<<<dim3(T_ / 32, H_, B_), dim3(64), 0, stream>>>(
        Qb, Kb, Vt, syb_mask, syb_graph, obuf);

    add_ln_kernel<<<dim3(M_), dim3(256), 0, stream>>>(
        obuf, xbuf, ln1_g + (size_t)i * D_, ln1_b + (size_t)i * D_, xbuf, xb);

    convT(ff_w1 + (size_t)i * D_ * FFH_, D_, FFH_, D_);
    gemm(xb, ff_b1 + (size_t)i * FFH_, nullptr, hbig, FFH_, D_, 1, 0);
    convT(ff_w2 + (size_t)i * FFH_ * D_, FFH_, D_, FFH_);
    gemm(hbig, ff_b2 + (size_t)i * D_, obuf, nullptr, D_, FFH_, 0, 0);

    add_ln_kernel<<<dim3(M_), dim3(256), 0, stream>>>(
        obuf, xbuf, ln2_g + (size_t)i * D_, ln2_b + (size_t)i * D_, xbuf, xb);
  }

  (void)hipMemcpyAsync(d_out, xbuf, (size_t)M_ * D_ * 4, hipMemcpyDeviceToDevice, stream);
}
